// MultiHeadAttention_74955769250530
// MI455X (gfx1250) — matrix operand order repaired and checked by reading, not compiled, not run on hardware
//
#include <hip/hip_runtime.h>
#include <hip/hip_bf16.h>

#define D_MODELc 1024
#define NUM_HEADSc 16
#define D_Hc 64
#define SEQc 2048
#define BATCHc 2

typedef __attribute__((ext_vector_type(16))) __bf16 v16bf;
typedef __attribute__((ext_vector_type(8)))  __bf16 v8bf;
typedef __attribute__((ext_vector_type(8)))  float  v8f;

static __device__ inline v8f zero8() {
    v8f z;
#pragma unroll
    for (int i = 0; i < 8; ++i) z[i] = 0.0f;
    return z;
}

static __device__ inline v16bf cat8(v8bf lo, v8bf hi) {
    return __builtin_shufflevector(lo, hi, 0,1,2,3,4,5,6,7,8,9,10,11,12,13,14,15);
}

static __device__ inline v8f wmma_bf16(v16bf a, v16bf b, v8f c) {
    // D = A(16x32 bf16) * B(32x16 bf16) + C(16x16 f32)
    return __builtin_amdgcn_wmma_f32_16x16x32_bf16(
        /*neg_a=*/false, a, /*neg_b=*/false, b,
        /*c_mod=*/(short)0, c, /*reuse_a=*/false, /*reuse_b=*/false);
}

// ---------------------------------------------------------------------------
// GEMM: C[M,1024] = A[M,1024] @ W[1024,1024]^T + bias
// A is f32 or bf16 (template), C is bf16 or f32 (template).
// Block = 128 threads (4 waves). Tile BM=64, BN=64, KT=32.
// Wave w owns rows w*16..w*16+15; 4 N-subtiles of 16 each.
// ---------------------------------------------------------------------------
template <bool A_IS_BF16, bool OUT_F32>
__global__ __launch_bounds__(128)
void gemm_wmma(const void* __restrict__ Av, const float* __restrict__ W,
               const float* __restrict__ bias, void* __restrict__ Cv) {
    const int tid  = threadIdx.x;
    const int lane = tid & 31;
    const int wv   = tid >> 5;
    const int m0   = blockIdx.x * 64;
    const int n0   = blockIdx.y * 64;
    const int hi   = lane >> 4;   // 0/1 : K-half selector
    const int lm   = lane & 15;

    __shared__ __bf16 aT[64 * 40];  // [row][k] padded stride 40 (80B, 16B-aligned rows)
    __shared__ __bf16 bT[64 * 40];  // [n][k]   = W[n0+n][kk+k]

    v8f acc[4];
#pragma unroll
    for (int s = 0; s < 4; ++s) acc[s] = zero8();

    const int sr = tid >> 1;          // staging row 0..63
    const int sc = (tid & 1) * 16;    // staging col 0 / 16

    for (int kk = 0; kk < D_MODELc; kk += 32) {
        // stage A tile (convert to bf16 if needed)
        if constexpr (A_IS_BF16) {
            const __bf16* src = (const __bf16*)Av + (size_t)(m0 + sr) * D_MODELc + kk + sc;
#pragma unroll
            for (int i = 0; i < 16; ++i) aT[sr * 40 + sc + i] = src[i];
        } else {
            const float* src = (const float*)Av + (size_t)(m0 + sr) * D_MODELc + kk + sc;
#pragma unroll
            for (int i = 0; i < 16; ++i) aT[sr * 40 + sc + i] = (__bf16)src[i];
        }
        // stage B tile from W (row n of W == column n of W^T : contiguous)
        {
            const float* src = W + (size_t)(n0 + sr) * D_MODELc + kk + sc;
#pragma unroll
            for (int i = 0; i < 16; ++i) bT[sr * 40 + sc + i] = (__bf16)src[i];
        }
        __syncthreads();

        // A fragment: lane row = wv*16 + lm; K elems {hi*8..+8} and {16+hi*8..+8}
        const __bf16* arow = aT + (wv * 16 + lm) * 40;
        v16bf af = cat8(*(const v8bf*)(arow + hi * 8),
                        *(const v8bf*)(arow + 16 + hi * 8));
#pragma unroll
        for (int s = 0; s < 4; ++s) {
            // B fragment: lane col N = s*16+lm; K elems {hi*8..+8} and {16+hi*8..+8}
            const __bf16* brow = bT + (s * 16 + lm) * 40 + hi * 8;
            v16bf bf = cat8(*(const v8bf*)brow, *(const v8bf*)(brow + 16));
            acc[s] = wmma_bf16(af, bf, acc[s]);
        }
        __syncthreads();
    }

    // write out: D element (M = hi*8+r, N = lm) per subtile
#pragma unroll
    for (int s = 0; s < 4; ++s) {
        const int ncol = n0 + s * 16 + lm;
        const float bval = bias[ncol];
#pragma unroll
        for (int r = 0; r < 8; ++r) {
            const int mrow = m0 + wv * 16 + hi * 8 + r;
            const float val = acc[s][r] + bval;
            if constexpr (OUT_F32)
                ((float*)Cv)[(size_t)mrow * D_MODELc + ncol] = val;
            else
                ((__bf16*)Cv)[(size_t)mrow * D_MODELc + ncol] = (__bf16)val;
        }
    }
}

// ---------------------------------------------------------------------------
// Fused relative attention per (b,h) row-block of 64 rows.
// Phase A: qet[0..64][0..2047] = q[i0..i0+64] @ E^T  (WMMA)  -> LDS (bf16)
// Phase B: flash attention over 64-wide column tiles:
//          S = qK^T (WMMA) + srel(gather from qet LDS), online softmax,
//          O += P @ V (WMMA, P re-laid out through per-wave LDS).
// Output written with the standard transpose: attn[b, s, h*64 + d] (bf16).
// ---------------------------------------------------------------------------
#define QET_E (65 * 2048)
#define QLD_E (64 * 72)
#define KLD_E (64 * 72)
#define VTL_E (64 * 72)
#define PLD_E (4 * 16 * 72)
#define ATTN_SMEM_BYTES ((QET_E + QLD_E + KLD_E + VTL_E + PLD_E) * 2)

__global__ __launch_bounds__(128)
void attn_wmma(const __bf16* __restrict__ qbuf, const __bf16* __restrict__ kbuf,
               const __bf16* __restrict__ vbuf, const float* __restrict__ E,
               __bf16* __restrict__ attn) {
    const int tid  = threadIdx.x;
    const int lane = tid & 31;
    const int wv   = tid >> 5;
    const int hi   = lane >> 4;
    const int lm   = lane & 15;

    const int i0 = blockIdx.x * 64;
    const int bh = blockIdx.y;          // 0..31
    const int b  = bh >> 4;
    const int h  = bh & 15;

    extern __shared__ char smem_raw[];
    __bf16* qet = (__bf16*)smem_raw;          // [65][2048]
    __bf16* qld = qet + QET_E;                // [64][72]
    __bf16* kld = qld + QLD_E;                // [64][72]
    __bf16* vtl = kld + KLD_E;                // [64][72]  (V transposed: [n][k])
    __bf16* pld = vtl + VTL_E;                // 4 x [16][72]

    const __bf16* qg = qbuf + (size_t)bh * SEQc * D_Hc;  // heads are contiguous (raw view!)
    const __bf16* kg = kbuf + (size_t)bh * SEQc * D_Hc;
    const __bf16* vg = vbuf + (size_t)bh * SEQc * D_Hc;

    // stage q rows i0..i0+63 into LDS (vectorized 16B copies)
    {
        const int r  = tid >> 1;
        const int c0 = (tid & 1) * 32;
        const v8bf* src = (const v8bf*)(qg + (size_t)(i0 + r) * D_Hc + c0);
        v8bf* dst = (v8bf*)(qld + r * 72 + c0);
#pragma unroll
        for (int i = 0; i < 4; ++i) dst[i] = src[i];
    }

    // ---------------- Phase A: qet = q(65 rows) @ E^T ----------------
    for (int t = wv; t < 5 * 128; t += 4) {
        const int mt = t / 128;   // 0..4 : 16-row tile (rows mt*16..+16, only 0..64 used)
        const int nt = t % 128;   // 0..127 : 16-col tile of the 2048 rel positions
        const int rr = mt * 16 + lm;  // A-row in block coords
        const bool okA = (rr <= 64) && (i0 + rr < SEQc);
        v16bf af[2];
        if (okA) {
            const __bf16* row = qg + (size_t)(i0 + rr) * D_Hc;
#pragma unroll
            for (int ks = 0; ks < 2; ++ks)
                af[ks] = cat8(*(const v8bf*)(row + ks * 32 + hi * 8),
                              *(const v8bf*)(row + ks * 32 + 16 + hi * 8));
        } else {
#pragma unroll
            for (int ks = 0; ks < 2; ++ks)
#pragma unroll
                for (int i = 0; i < 16; ++i) af[ks][i] = (__bf16)0.0f;
        }
        v8f d = zero8();
#pragma unroll
        for (int ks = 0; ks < 2; ++ks) {
            // B col N = nt*16+lm  ==  E row (nt*16+lm); K elems ks*32 + {hi*8..+8, 16+hi*8..+8}
            const float* erow = E + (size_t)(nt * 16 + lm) * D_Hc + ks * 32 + hi * 8;
            v16bf bf;
#pragma unroll
            for (int i = 0; i < 16; ++i) bf[i] = (__bf16)erow[i < 8 ? i : i + 8];
            d = wmma_bf16(af[ks], bf, d);
        }
#pragma unroll
        for (int r = 0; r < 8; ++r) {
            const int m = mt * 16 + hi * 8 + r;
            if (m <= 64) qet[m * 2048 + nt * 16 + lm] = (__bf16)d[r];
        }
    }
    __syncthreads();

    // ---------------- Phase B: flash attention ----------------
    float runmax[8], lsum[8];
#pragma unroll
    for (int r = 0; r < 8; ++r) { runmax[r] = -3.0e38f; lsum[r] = 0.0f; }
    v8f outa[4];
#pragma unroll
    for (int s = 0; s < 4; ++s) outa[s] = zero8();

    __bf16* myp = pld + wv * 16 * 72;

    for (int j0 = 0; j0 < SEQc; j0 += 64) {
        __syncthreads();  // previous tile fully consumed
        {
            const int r  = tid >> 1;
            const int c0 = (tid & 1) * 32;
            const v8bf* ksrc = (const v8bf*)(kg + (size_t)(j0 + r) * D_Hc + c0);
            v8bf* kdst = (v8bf*)(kld + r * 72 + c0);
#pragma unroll
            for (int i = 0; i < 4; ++i) kdst[i] = ksrc[i];
            // V transposed on store: vtl[n][k] = V[j0+k][n]
            const __bf16* vsrc = vg + (size_t)(j0 + r) * D_Hc + c0;
#pragma unroll
            for (int i = 0; i < 32; ++i) vtl[(c0 + i) * 72 + r] = vsrc[i];
        }
        __syncthreads();

        // S = q @ K^T
        v8f sacc[4];
#pragma unroll
        for (int s = 0; s < 4; ++s) sacc[s] = zero8();
#pragma unroll
        for (int ks = 0; ks < 2; ++ks) {
            const __bf16* arow = qld + (wv * 16 + lm) * 72 + ks * 32;
            v16bf af = cat8(*(const v8bf*)(arow + hi * 8),
                            *(const v8bf*)(arow + 16 + hi * 8));
#pragma unroll
            for (int s = 0; s < 4; ++s) {
                const __bf16* brow = kld + (s * 16 + lm) * 72 + ks * 32 + hi * 8;
                v16bf bf = cat8(*(const v8bf*)brow, *(const v8bf*)(brow + 16));
                sacc[s] = wmma_bf16(af, bf, sacc[s]);
            }
        }

        // scores = (S + srel) / 8 with exact skew semantics; online softmax
        float sc[4][8];
        float tmax[8];
#pragma unroll
        for (int r = 0; r < 8; ++r) tmax[r] = -3.0e38f;
#pragma unroll
        for (int s = 0; s < 4; ++s) {
            const int j = j0 + s * 16 + lm;
#pragma unroll
            for (int r = 0; r < 8; ++r) {
                const int i = i0 + wv * 16 + hi * 8 + r;
                float srel;
                if (j <= i)
                    srel = (float)qet[(i - i0) * 2048 + (SEQc - 1 + j - i)];
                else if (j == i + 1)
                    srel = -__builtin_inff();
                else
                    srel = (float)qet[(i + 1 - i0) * 2048 + (j - i - 2)];
                const float v = (sacc[s][r] + srel) * 0.125f;
                sc[s][r] = v;
                tmax[r] = fmaxf(tmax[r], v);
            }
        }
        // row reductions across the 16 lanes sharing this row group
#pragma unroll
        for (int r = 0; r < 8; ++r) {
#pragma unroll
            for (int m = 1; m < 16; m <<= 1)
                tmax[r] = fmaxf(tmax[r], __shfl_xor(tmax[r], m, 32));
        }
        float scl[8], tsum[8];
#pragma unroll
        for (int r = 0; r < 8; ++r) {
            const float nm = fmaxf(runmax[r], tmax[r]);
            scl[r] = __expf(runmax[r] - nm);
            runmax[r] = nm;
            tsum[r] = 0.0f;
        }
#pragma unroll
        for (int s = 0; s < 4; ++s) {
#pragma unroll
            for (int r = 0; r < 8; ++r) {
                const float p = __expf(sc[s][r] - runmax[r]);
                tsum[r] += p;
                myp[(hi * 8 + r) * 72 + s * 16 + lm] = (__bf16)p;  // D-layout -> row-major
            }
        }
#pragma unroll
        for (int r = 0; r < 8; ++r) {
#pragma unroll
            for (int m = 1; m < 16; m <<= 1)
                tsum[r] += __shfl_xor(tsum[r], m, 32);
            lsum[r] = lsum[r] * scl[r] + tsum[r];
        }
#pragma unroll
        for (int s = 0; s < 4; ++s)
#pragma unroll
            for (int r = 0; r < 8; ++r) outa[s][r] *= scl[r];

        // O += P @ V  (A from per-wave LDS, B from transposed V)
#pragma unroll
        for (int ks = 0; ks < 2; ++ks) {
            const __bf16* arow = myp + lm * 72 + ks * 32;
            v16bf af = cat8(*(const v8bf*)(arow + hi * 8),
                            *(const v8bf*)(arow + 16 + hi * 8));
#pragma unroll
            for (int s = 0; s < 4; ++s) {
                const __bf16* brow = vtl + (s * 16 + lm) * 72 + ks * 32 + hi * 8;
                v16bf bf = cat8(*(const v8bf*)brow, *(const v8bf*)(brow + 16));
                outa[s] = wmma_bf16(af, bf, outa[s]);
            }
        }
    }

    // finalize: divide by softmax denom, write with standard [B,S,H*dh] transpose
#pragma unroll
    for (int s = 0; s < 4; ++s) {
#pragma unroll
        for (int r = 0; r < 8; ++r) {
            const int srow = i0 + wv * 16 + hi * 8 + r;
            const int ncol = s * 16 + lm;
            const float o = outa[s][r] / lsum[r];
            attn[((size_t)(b * SEQc + srow)) * D_MODELc + h * D_Hc + ncol] = (__bf16)o;
        }
    }
}

// ---------------------------------------------------------------------------
extern "C" void kernel_launch(void* const* d_in, const int* in_sizes, int n_in,
                              void* d_out, int out_size, void* d_ws, size_t ws_size,
                              hipStream_t stream) {
    (void)in_sizes; (void)n_in; (void)out_size; (void)ws_size;

    const float* Q  = (const float*)d_in[0];
    const float* K  = (const float*)d_in[1];
    const float* V  = (const float*)d_in[2];
    const float* Wq = (const float*)d_in[3];
    const float* bq = (const float*)d_in[4];
    const float* Wk = (const float*)d_in[5];
    const float* bk = (const float*)d_in[6];
    const float* Wv = (const float*)d_in[7];
    const float* bv = (const float*)d_in[8];
    const float* E  = (const float*)d_in[9];
    const float* Wo = (const float*)d_in[10];
    const float* bo = (const float*)d_in[11];

    const int M = BATCHc * SEQc;  // 4096
    __bf16* qb = (__bf16*)d_ws;
    __bf16* kb = qb + (size_t)M * D_MODELc;
    __bf16* vb = kb + (size_t)M * D_MODELc;
    __bf16* ab = vb + (size_t)M * D_MODELc;

    dim3 gg(M / 64, D_MODELc / 64);
    dim3 blk(128);
    gemm_wmma<false, false><<<gg, blk, 0, stream>>>((const void*)Q, Wq, bq, (void*)qb);
    gemm_wmma<false, false><<<gg, blk, 0, stream>>>((const void*)K, Wk, bk, (void*)kb);
    gemm_wmma<false, false><<<gg, blk, 0, stream>>>((const void*)V, Wv, bv, (void*)vb);

    dim3 ga(SEQc / 64, BATCHc * NUM_HEADSc);
    attn_wmma<<<ga, blk, ATTN_SMEM_BYTES, stream>>>(qb, kb, vb, E, ab);

    gemm_wmma<true, true><<<gg, blk, 0, stream>>>((const void*)ab, Wo, bo, d_out);
}
